// GATModel_10299331576448
// MI455X (gfx1250) — compile-verified, matrix operand order checked by reading, not run on hardware
//
#include <hip/hip_runtime.h>
#include <hip/hip_bf16.h>

// ---------------------------------------------------------------------------
// Types for CDNA5 WMMA (gfx1250, wave32)
// ---------------------------------------------------------------------------
typedef __attribute__((ext_vector_type(16))) __bf16 v16bf;
typedef __attribute__((ext_vector_type(8)))  __bf16 v8bf;
typedef __attribute__((ext_vector_type(8)))  float  v8f;
typedef __attribute__((ext_vector_type(4)))  int    v4i;

#define G7 7
#define NNODE 49           // 7x7
#define BATCH 256
#define NN_TOT (BATCH * NNODE)   // 12544
#define F_IN 1024
#define H1 8
#define C1 128
#define HC1 1024
#define O2 256
#define KNN_K 3
#define NEG_SLOPE 0.2f

// CDNA5 async global->LDS copy availability (device pass only)
#if defined(__gfx1250__) && __has_builtin(__builtin_amdgcn_global_load_async_to_lds_b128)
#define HAVE_ASYNC_LDS 1
#else
#define HAVE_ASYNC_LDS 0
#endif

#if HAVE_ASYNC_LDS
// builtin signature (from hipcc diagnostic): (int4 addrspace(1)*, int4 addrspace(3)*, imm offset, imm cpol)
#define ASYNC_CP16(gsrc, ldst)                                               \
  __builtin_amdgcn_global_load_async_to_lds_b128(                            \
      (__attribute__((address_space(1))) v4i*)(gsrc),                        \
      (__attribute__((address_space(3))) v4i*)(ldst), 0, 0)
#if __has_builtin(__builtin_amdgcn_s_wait_asynccnt)
#define WAIT_ASYNC(n) __builtin_amdgcn_s_wait_asynccnt(n)
#else
#define WAIT_ASYNC(n) asm volatile("s_wait_asynccnt %0" ::"n"(n) : "memory")
#endif
#endif

// ---------------------------------------------------------------------------
// Kernel 1: relu(feat_map) [B, F, 49] -> xb [B*49, F] bf16
// ---------------------------------------------------------------------------
__global__ void k_prep_x(const float* __restrict__ fm, __bf16* __restrict__ xb,
                         int total) {
  int idx = blockIdx.x * blockDim.x + threadIdx.x;
  if (idx >= total) return;
  int f = idx & (F_IN - 1);
  int bn = idx >> 10;           // F_IN == 1024
  int n = bn % NNODE;
  int b = bn / NNODE;
  float v = fm[((size_t)b * F_IN + f) * NNODE + n];
  xb[idx] = (__bf16)fmaxf(v, 0.0f);
}

// ---------------------------------------------------------------------------
// Kernel 2: transpose W [K, Nc] f32 -> Wt [Nc, K] bf16
// ---------------------------------------------------------------------------
__global__ void k_wt(const float* __restrict__ W, __bf16* __restrict__ Wt,
                     int K, int Nc) {
  int idx = blockIdx.x * blockDim.x + threadIdx.x;
  if (idx >= K * Nc) return;
  int k = idx % K;
  int n = idx / K;
  Wt[idx] = (__bf16)W[(size_t)k * Nc + n];
}

// ---------------------------------------------------------------------------
// Kernel 3: tiled bf16 WMMA GEMM.  C[M,N] = A[M,K] * Bt[N,K]^T  (f32 accum)
//   block = 256 threads = 8 wave32, tile 128x128, K-step 32
//   per wave: 32x64 -> 2 A frags x 4 B frags x v8f acc, 8 WMMAs / k-step
//   global->LDS via async copy (ASYNCcnt) + double-buffered LDS pipeline
// ---------------------------------------------------------------------------
#define TM 128
#define TN 128
#define TK 32
#define LDT (TK + 8)   // pad to keep 16B alignment, dodge bank conflicts

__global__ __launch_bounds__(256) void k_gemm_bf16(
    const __bf16* __restrict__ A,   // [M, K] row-major
    const __bf16* __restrict__ Bt,  // [N, K] row-major (B transposed)
    float* __restrict__ C,          // [M, N]
    int M, int N, int K) {
  __shared__ __bf16 As[2][TM][LDT];
  __shared__ __bf16 Bs[2][TN][LDT];

  const int tid  = threadIdx.x;
  const int lane = tid & 31;
  const int wave = tid >> 5;
  const int wm   = wave >> 1;          // 0..3  (M direction, 32 rows each)
  const int wn   = wave & 1;           // 0..1  (N direction, 64 cols each)
  const int m0   = blockIdx.x * TM;
  const int n0   = blockIdx.y * TN;
  const int lm   = lane & 15;
  const int hsel = (lane & 16) ? 8 : 0;   // K phase per 16-bit A/B layout

  v8f acc[2][4];
#pragma unroll
  for (int i = 0; i < 2; ++i)
#pragma unroll
    for (int j = 0; j < 4; ++j)
#pragma unroll
      for (int e = 0; e < 8; ++e) acc[i][j][e] = 0.0f;

  const int row  = tid >> 1;          // 0..127 tile row loaded by this thread
  const int colh = (tid & 1) * 16;    // first or second 16-element half
  const __bf16* gA = A  + (size_t)(m0 + row) * K + colh;
  const __bf16* gB = Bt + (size_t)(n0 + row) * K + colh;
  const int nstage = K / TK;

#if HAVE_ASYNC_LDS
  // ---- prologue: stage 0 into buffer 0 (4 async b128 per thread) ----
  ASYNC_CP16(gA,     &As[0][row][colh]);
  ASYNC_CP16(gA + 8, &As[0][row][colh + 8]);
  ASYNC_CP16(gB,     &Bs[0][row][colh]);
  ASYNC_CP16(gB + 8, &Bs[0][row][colh + 8]);
#endif

  for (int s = 0; s < nstage; ++s) {
    const int buf = s & 1;
#if HAVE_ASYNC_LDS
    if (s + 1 < nstage) {
      const __bf16* na = gA + (s + 1) * TK;
      const __bf16* nb = gB + (s + 1) * TK;
      ASYNC_CP16(na,     &As[buf ^ 1][row][colh]);
      ASYNC_CP16(na + 8, &As[buf ^ 1][row][colh + 8]);
      ASYNC_CP16(nb,     &Bs[buf ^ 1][row][colh]);
      ASYNC_CP16(nb + 8, &Bs[buf ^ 1][row][colh + 8]);
      WAIT_ASYNC(4);   // async loads complete in order: stage s is resident
    } else {
      WAIT_ASYNC(0);
    }
    if (s + 2 < nstage) {  // keep L2 warm two slabs ahead
      __builtin_prefetch(gA + (s + 2) * TK, 0, 1);
      __builtin_prefetch(gB + (s + 2) * TK, 0, 1);
    }
#else
    {  // fallback: batched register staging (issue all loads, one wait)
      const v8bf* ga = (const v8bf*)(gA + s * TK);
      const v8bf* gb = (const v8bf*)(gB + s * TK);
      v8bf ta0 = ga[0], ta1 = ga[1], tb0 = gb[0], tb1 = gb[1];
      *(v8bf*)&As[buf][row][colh]     = ta0;
      *(v8bf*)&As[buf][row][colh + 8] = ta1;
      *(v8bf*)&Bs[buf][row][colh]     = tb0;
      *(v8bf*)&Bs[buf][row][colh + 8] = tb1;
    }
#endif
    __syncthreads();

    // ---- fragments: lane<16 holds K {0..7,16..23}, lane>=16 {8..15,24..31}
    v16bf afrag[2], bfrag[4];
#pragma unroll
    for (int i = 0; i < 2; ++i) {
      const __bf16* p = &As[buf][wm * 32 + i * 16 + lm][hsel];
      ((v8bf*)&afrag[i])[0] = *(const v8bf*)p;
      ((v8bf*)&afrag[i])[1] = *(const v8bf*)(p + 16);
    }
#pragma unroll
    for (int j = 0; j < 4; ++j) {
      const __bf16* p = &Bs[buf][wn * 64 + j * 16 + lm][hsel];
      ((v8bf*)&bfrag[j])[0] = *(const v8bf*)p;
      ((v8bf*)&bfrag[j])[1] = *(const v8bf*)(p + 16);
    }
#pragma unroll
    for (int i = 0; i < 2; ++i)
#pragma unroll
      for (int j = 0; j < 4; ++j)
        acc[i][j] = __builtin_amdgcn_wmma_f32_16x16x32_bf16(
            false, afrag[i], false, bfrag[j], (short)0, acc[i][j],
            false, false);
    __syncthreads();   // all waves done reading buf before it is refilled
  }

  // ---- store C: lane<16 -> rows r, lane>=16 -> rows r+8; N = lane&15 ----
  const int rbase = (lane & 16) ? 8 : 0;
#pragma unroll
  for (int i = 0; i < 2; ++i)
#pragma unroll
    for (int j = 0; j < 4; ++j)
#pragma unroll
      for (int r = 0; r < 8; ++r)
        C[(size_t)(m0 + wm * 32 + i * 16 + rbase + r) * N +
          (n0 + wn * 64 + j * 16 + lm)] = acc[i][j][r];
}

// ---------------------------------------------------------------------------
// Kernel 4: per-graph kNN (k=3) in feature space, from bf16 features
// ---------------------------------------------------------------------------
__global__ __launch_bounds__(256) void k_knn(const __bf16* __restrict__ xb,
                                             int* __restrict__ knn) {
  const int g = blockIdx.x;
  const int t = threadIdx.x;
  __shared__ float sq[NNODE];
  __shared__ float dist[NNODE * NNODE];
  const __bf16* X = xb + (size_t)g * NNODE * F_IN;

  if (t < NNODE) {
    float s = 0.0f;
    for (int k = 0; k < F_IN; k += 8) {
      v8bf v = *(const v8bf*)(X + (size_t)t * F_IN + k);
#pragma unroll
      for (int e = 0; e < 8; ++e) { float f = (float)v[e]; s += f * f; }
    }
    sq[t] = s;
  }
  __syncthreads();

  for (int p = t; p < NNODE * NNODE; p += 256) {
    int i = p / NNODE, j = p % NNODE;
    float dot = 0.0f;
    for (int k = 0; k < F_IN; k += 8) {
      v8bf a = *(const v8bf*)(X + (size_t)i * F_IN + k);
      v8bf b = *(const v8bf*)(X + (size_t)j * F_IN + k);
#pragma unroll
      for (int e = 0; e < 8; ++e) dot += (float)a[e] * (float)b[e];
    }
    float d = sq[i] + sq[j] - 2.0f * dot;
    if (i == j) d += 1e9f;
    dist[p] = d;
  }
  __syncthreads();

  if (t < NNODE) {
    float d0 = 1e30f, d1 = 1e30f, d2 = 1e30f;
    int b0 = 0, b1 = 0, b2 = 0;
    for (int j = 0; j < NNODE; ++j) {
      float d = dist[t * NNODE + j];
      if (d < d0)      { d2 = d1; b2 = b1; d1 = d0; b1 = b0; d0 = d; b0 = j; }
      else if (d < d1) { d2 = d1; b2 = b1; d1 = d; b1 = j; }
      else if (d < d2) { d2 = d; b2 = j; }
    }
    int base = (g * NNODE + t) * KNN_K;
    knn[base + 0] = b0; knn[base + 1] = b1; knn[base + 2] = b2;
  }
}

// ---------------------------------------------------------------------------
// Kernel 5: attention coefficients es/ed = <h, a_src>, <h, a_dst> per head
// ---------------------------------------------------------------------------
__global__ void k_att_coef(const float* __restrict__ h,
                           const float* __restrict__ a_src,
                           const float* __restrict__ a_dst,
                           float* __restrict__ es, float* __restrict__ ed,
                           int H, int C, int total /* NN*H */) {
  int idx = blockIdx.x * blockDim.x + threadIdx.x;
  if (idx >= total) return;
  int node = idx / H, head = idx % H;
  const float* hp = h + (size_t)node * H * C + head * C;
  float s = 0.0f, d = 0.0f;
  for (int c = 0; c < C; ++c) {
    float v = hp[c];
    s += v * a_src[head * C + c];
    d += v * a_dst[head * C + c];
  }
  es[idx] = s;
  ed[idx] = d;
}

// ---------------------------------------------------------------------------
// Kernel 6: GAT softmax-aggregate (one wave32 per (node, head)),
//           in-edges = 8-neighborhood + kNN(3) + self (<=12). +bias, ELU.
// ---------------------------------------------------------------------------
__global__ __launch_bounds__(256) void k_gat_agg(
    const float* __restrict__ h,     // [NN, H*C]
    const float* __restrict__ es,    // [NN, H]
    const float* __restrict__ ed,    // [NN, H]
    const int* __restrict__ knn,     // [NN, 3]
    const float* __restrict__ bias,  // [H*C]
    float* __restrict__ outf,        // optional f32 out
    __bf16* __restrict__ outb,       // optional bf16 out
    int H, int C, int total /* NN*H */) {
  int wavei = (blockIdx.x * blockDim.x + threadIdx.x) >> 5;
  int lane = threadIdx.x & 31;
  if (wavei >= total) return;
  int node = wavei / H, head = wavei % H;
  int g = node / NNODE, i = node % NNODE;
  int r = i / G7, c = i % G7;

  int srcs[12];
  int ne = 0;
#pragma unroll
  for (int dr = -1; dr <= 1; ++dr)
#pragma unroll
    for (int dc = -1; dc <= 1; ++dc) {
      if (dr == 0 && dc == 0) continue;
      int rr = r + dr, cc = c + dc;
      if (rr >= 0 && rr < G7 && cc >= 0 && cc < G7) srcs[ne++] = rr * G7 + cc;
    }
#pragma unroll
  for (int q = 0; q < KNN_K; ++q) srcs[ne++] = knn[node * KNN_K + q];
  srcs[ne++] = i;  // self loop

  const int base = g * NNODE;
  const float edi = ed[node * H + head];
  float ev[12];
  float m = -1e30f;
  for (int q = 0; q < ne; ++q) {
    float v = es[(base + srcs[q]) * H + head] + edi;
    v = v > 0.0f ? v : NEG_SLOPE * v;        // leaky_relu
    ev[q] = v;
    m = fmaxf(m, v);
  }
  float denom = 0.0f;
  for (int q = 0; q < ne; ++q) { ev[q] = __expf(ev[q] - m); denom += ev[q]; }
  const float inv = 1.0f / fmaxf(denom, 1e-16f);

  const int nd = C >> 5;                     // dims per lane (<= 8)
  float acc[8];
#pragma unroll
  for (int u = 0; u < 8; ++u) acc[u] = 0.0f;
  for (int q = 0; q < ne; ++q) {
    const float* hp = h + (size_t)(base + srcs[q]) * H * C + head * C;
    float a = ev[q] * inv;
    for (int u = 0; u < nd; ++u) acc[u] += a * hp[lane + u * 32];
  }
  for (int u = 0; u < nd; ++u) {
    int ci = lane + u * 32;
    float v = acc[u] + bias[head * C + ci];
    v = v > 0.0f ? v : (__expf(v) - 1.0f);   // ELU
    size_t oidx = (size_t)node * H * C + head * C + ci;
    if (outf) outf[oidx] = v;
    if (outb) outb[oidx] = (__bf16)v;
  }
}

// ---------------------------------------------------------------------------
// Kernel 7: global mean pool over 49 nodes -> pooled [B, 256]
// ---------------------------------------------------------------------------
__global__ __launch_bounds__(256) void k_pool(const float* __restrict__ h2,
                                              float* __restrict__ pooled) {
  int g = blockIdx.x, cch = threadIdx.x;
  float s = 0.0f;
  for (int n = 0; n < NNODE; ++n)
    s += h2[(size_t)(g * NNODE + n) * O2 + cch];
  pooled[g * O2 + cch] = s * (1.0f / (float)NNODE);
}

// ---------------------------------------------------------------------------
// Kernel 8: classifier out = relu(pooled@Wc1+bc1)@Wc2+bc2  -> [B, 14]
// ---------------------------------------------------------------------------
__global__ __launch_bounds__(256) void k_classifier(
    const float* __restrict__ pooled, const float* __restrict__ Wc1,
    const float* __restrict__ bc1, const float* __restrict__ Wc2,
    const float* __restrict__ bc2, float* __restrict__ out) {
  __shared__ float t[O2];
  int b = blockIdx.x, j = threadIdx.x;
  float acc = bc1[j];
  for (int k = 0; k < O2; ++k) acc += pooled[b * O2 + k] * Wc1[k * O2 + j];
  t[j] = fmaxf(acc, 0.0f);
  __syncthreads();
  if (j < 14) {
    float o = bc2[j];
    for (int k = 0; k < O2; ++k) o += t[k] * Wc2[k * 14 + j];
    out[b * 14 + j] = o;
  }
}

// ---------------------------------------------------------------------------
// Host-side launcher
// ---------------------------------------------------------------------------
extern "C" void kernel_launch(void* const* d_in, const int* in_sizes, int n_in,
                              void* d_out, int out_size, void* d_ws,
                              size_t ws_size, hipStream_t stream) {
  (void)in_sizes; (void)n_in; (void)out_size; (void)ws_size;

  const float* feat_map = (const float*)d_in[0];   // [256,1024,7,7]
  const float* W1       = (const float*)d_in[1];   // [1024,1024]
  const float* a_src1   = (const float*)d_in[2];   // [8,128]
  const float* a_dst1   = (const float*)d_in[3];
  const float* b1       = (const float*)d_in[4];   // [1024]
  const float* W2       = (const float*)d_in[5];   // [1024,256]
  const float* a_src2   = (const float*)d_in[6];   // [1,256]
  const float* a_dst2   = (const float*)d_in[7];
  const float* b2       = (const float*)d_in[8];   // [256]
  const float* Wc1      = (const float*)d_in[9];   // [256,256]
  const float* bc1      = (const float*)d_in[10];
  const float* Wc2      = (const float*)d_in[11];  // [256,14]
  const float* bc2      = (const float*)d_in[12];
  float* out = (float*)d_out;

  // -------- workspace carve-out (256B aligned) --------
  char* ws = (char*)d_ws;
  size_t off = 0;
  auto alloc = [&](size_t bytes) -> void* {
    off = (off + 255) & ~(size_t)255;
    void* p = ws + off;
    off += bytes;
    return p;
  };
  __bf16* xb   = (__bf16*)alloc((size_t)NN_TOT * F_IN * 2);
  __bf16* W1t  = (__bf16*)alloc((size_t)HC1 * F_IN * 2);      // [1024,1024]
  __bf16* W2t  = (__bf16*)alloc((size_t)O2 * HC1 * 2);        // [256,1024]
  float*  h1   = (float*)alloc((size_t)NN_TOT * HC1 * 4);
  __bf16* h1b  = (__bf16*)alloc((size_t)NN_TOT * HC1 * 2);
  float*  z2   = (float*)alloc((size_t)NN_TOT * O2 * 4);
  float*  h2f  = (float*)alloc((size_t)NN_TOT * O2 * 4);
  float*  es1  = (float*)alloc((size_t)NN_TOT * H1 * 4);
  float*  ed1  = (float*)alloc((size_t)NN_TOT * H1 * 4);
  float*  es2  = (float*)alloc((size_t)NN_TOT * 4);
  float*  ed2  = (float*)alloc((size_t)NN_TOT * 4);
  int*    knn  = (int*)alloc((size_t)NN_TOT * KNN_K * 4);
  float*  pooled = (float*)alloc((size_t)BATCH * O2 * 4);

  // 1) features: relu + transpose + bf16
  {
    int total = NN_TOT * F_IN;
    k_prep_x<<<(total + 255) / 256, 256, 0, stream>>>(feat_map, xb, total);
  }
  // 2) weight transposes to [N,K] bf16
  k_wt<<<(F_IN * HC1 + 255) / 256, 256, 0, stream>>>(W1, W1t, F_IN, HC1);
  k_wt<<<(HC1 * O2 + 255) / 256, 256, 0, stream>>>(W2, W2t, HC1, O2);
  // 3) kNN edges (built once from backbone features, used by both layers)
  k_knn<<<BATCH, 256, 0, stream>>>(xb, knn);
  // 4) GEMM1: h1 = xb @ W1   (12544 x 1024 x 1024, WMMA bf16)
  k_gemm_bf16<<<dim3(NN_TOT / TM, HC1 / TN), 256, 0, stream>>>(
      xb, W1t, h1, NN_TOT, HC1, F_IN);
  // 5) layer-1 attention coefficients + aggregate (+b1, ELU) -> bf16
  {
    int total = NN_TOT * H1;
    k_att_coef<<<(total + 255) / 256, 256, 0, stream>>>(h1, a_src1, a_dst1,
                                                        es1, ed1, H1, C1, total);
    k_gat_agg<<<(total + 7) / 8, 256, 0, stream>>>(
        h1, es1, ed1, knn, b1, (float*)nullptr, h1b, H1, C1, total);
  }
  // 6) GEMM2: z2 = h1b @ W2  (12544 x 1024 x 256, WMMA bf16)
  k_gemm_bf16<<<dim3(NN_TOT / TM, O2 / TN), 256, 0, stream>>>(
      h1b, W2t, z2, NN_TOT, O2, HC1);
  // 7) layer-2 attention (+b2, ELU) -> f32
  {
    int total = NN_TOT;  // H = 1
    k_att_coef<<<(total + 255) / 256, 256, 0, stream>>>(z2, a_src2, a_dst2,
                                                        es2, ed2, 1, O2, total);
    k_gat_agg<<<(total + 7) / 8, 256, 0, stream>>>(
        z2, es2, ed2, knn, b2, h2f, (__bf16*)nullptr, 1, O2, total);
  }
  // 8) mean pool + classifier
  k_pool<<<BATCH, 256, 0, stream>>>(h2f, pooled);
  k_classifier<<<BATCH, 256, 0, stream>>>(pooled, Wc1, bc1, Wc2, bc2, out);
}
